// DecoderRNNT_12549894439370
// MI455X (gfx1250) — compile-verified, matrix operand order checked by reading, not run on hardware
//
#include <hip/hip_runtime.h>

// ---------------------------------------------------------------------------
// RNN-T decoder + joint for MI455X (gfx1250, wave32, WMMA, TDM).
// Dominant cost: fused tanh-joint GEMM, output 268 MB -> ~11.5us HBM floor.
// ---------------------------------------------------------------------------

#define BB   4
#define TT   256
#define UU   64
#define DU   512
#define GG   2048   // 4*DU
#define JJ   512
#define OD   1024
#define EMB  512
#define KK   512

typedef __attribute__((ext_vector_type(16))) _Float16 v16h;
typedef __attribute__((ext_vector_type(8)))  float    v8f;
typedef __attribute__((ext_vector_type(2)))  float    v2f;
typedef __attribute__((ext_vector_type(4)))  unsigned int v4u;
typedef __attribute__((ext_vector_type(8)))  int      v8i;
typedef __attribute__((ext_vector_type(4)))  int      v4i;

#ifdef __has_builtin
#if __has_builtin(__builtin_amdgcn_wmma_f32_16x16x4_f32)
#define HAVE_WMMA_F32F32 1
#endif
#if __has_builtin(__builtin_amdgcn_tensor_load_to_lds)
#define HAVE_TDM 1
#endif
#endif

// A/B fragment loaders for V_WMMA_*_16X16X32_F16 (wave32):
// lane L holds row (L&15); elems 0..7 = K[kb..kb+7], elems 8..15 = K[kb+16..kb+23]
// with kb = kstep*32 + 8*(L>>4).  (ISA 7.12.2, 16-bit 16x32 layout.)
static __device__ inline v16h frag_from_f32(const float* __restrict__ p) {
  v16h r;
#pragma unroll
  for (int i = 0; i < 8; ++i) { r[i] = (_Float16)p[i]; r[i + 8] = (_Float16)p[i + 16]; }
  return r;
}
static __device__ inline v16h frag_from_f16(const _Float16* __restrict__ p) {
  v16h r;
#pragma unroll
  for (int i = 0; i < 8; ++i) { r[i] = p[i]; r[i + 8] = p[i + 16]; }
  return r;
}
static __device__ inline float sigf(float x) { return 1.0f / (1.0f + __expf(-x)); }

// ---------------------------------------------------------------------------
// Kernel 1: G0x[b,u,:] = embed[ys_in[b,u]] @ W_ih0^T + b_ih0 + b_hh0
// Embedding gather fused into the WMMA A-operand loader. M=256, N=2048, K=512.
// Dual accumulators break the D->C WMMA hazard chain (ISA 7.12.1).
// ---------------------------------------------------------------------------
__global__ void __launch_bounds__(256) k_gates0(
    const float* __restrict__ embed, const int* __restrict__ ys,
    const float* __restrict__ Wih0, const float* __restrict__ bih0,
    const float* __restrict__ bhh0, float* __restrict__ G0x) {
  const int tid = threadIdx.x, lane = tid & 31, wave = tid >> 5;
  const int tile = blockIdx.x * 8 + wave;          // 16 Mtiles x 128 Ntiles
  const int mt = tile >> 7, nt = tile & 127;
  const int m0 = mt * 16, n0 = nt * 16;
  const int hh = lane >> 4, ln = lane & 15;

  const int m = m0 + ln;                           // A-row for this lane
  const int b = m >> 6, u = m & 63;
  const int row = (u == 0) ? 0 : ys[b * 63 + (u - 1)];  // blank row is zeroed
  const float* arow = embed + (size_t)row * EMB;
  const float* brow = Wih0 + (size_t)(n0 + ln) * KK;    // W is N x K row-major

  v8f a0 = {}, a1 = {};
#pragma unroll 4
  for (int ks = 0; ks < 16; ks += 2) {
    const int kb0 = ks * 32 + 8 * hh, kb1 = kb0 + 32;
    v16h x0 = frag_from_f32(arow + kb0);
    v16h w0 = frag_from_f32(brow + kb0);
    v16h x1 = frag_from_f32(arow + kb1);
    v16h w1 = frag_from_f32(brow + kb1);
    a0 = __builtin_amdgcn_wmma_f32_16x16x32_f16(false, x0, false, w0, (short)0, a0, false, false);
    a1 = __builtin_amdgcn_wmma_f32_16x16x32_f16(false, x1, false, w1, (short)0, a1, false, false);
  }
  const v8f acc = a0 + a1;
  const int n = n0 + ln;
  const float bias = bih0[n] + bhh0[n];
#pragma unroll
  for (int v = 0; v < 8; ++v)
    G0x[(size_t)(m0 + v + 8 * hh) * GG + n] = acc[v] + bias;
}

// ---------------------------------------------------------------------------
// Kernel 2: generic fp32 GEMM  C[M,512] = A[M,512] @ W[512,512]^T (+bias)
// Uses V_WMMA_F32_16X16X4_F32 (signature confirmed by round-1 codegen).
// ---------------------------------------------------------------------------
__global__ void __launch_bounds__(256) k_gemm_f32(
    const float* __restrict__ A, const float* __restrict__ W,
    const float* __restrict__ bias, float* __restrict__ C, int M) {
  const int tid = threadIdx.x, lane = tid & 31, wave = tid >> 5;
  const int tile = blockIdx.x * 8 + wave;          // (M/16) x 32 tiles
  const int mt = tile >> 5, nt = tile & 31;
  const int m0 = mt * 16, n0 = nt * 16;
  const int hh = lane >> 4, ln = lane & 15;
  (void)M;
  v8f a0 = {}, a1 = {};
#ifdef HAVE_WMMA_F32F32
  // f32 A 16x4: VGPR0 = K{0|2}, VGPR1 = K{1|3} -> lane loads float2 at k0+2*hh
  const float* arow = A + (size_t)(m0 + ln) * KK;
  const float* brow = W + (size_t)(n0 + ln) * KK;
#pragma unroll 8
  for (int ks = 0; ks < 128; ks += 2) {
    const int kb0 = ks * 4 + 2 * hh, kb1 = kb0 + 4;
    v2f x0 = *(const v2f*)(arow + kb0);
    v2f w0 = *(const v2f*)(brow + kb0);
    v2f x1 = *(const v2f*)(arow + kb1);
    v2f w1 = *(const v2f*)(brow + kb1);
    a0 = __builtin_amdgcn_wmma_f32_16x16x4_f32(false, x0, false, w0, (short)0, a0, false, false);
    a1 = __builtin_amdgcn_wmma_f32_16x16x4_f32(false, x1, false, w1, (short)0, a1, false, false);
  }
#else
  // VALU fallback producing the same per-lane C layout.
  const float* brow = W + (size_t)(n0 + ln) * KK;
#pragma unroll
  for (int v = 0; v < 8; ++v) {
    const float* ar = A + (size_t)(m0 + v + 8 * hh) * KK;
    float s = 0.f;
    for (int k = 0; k < KK; ++k) s += ar[k] * brow[k];
    a0[v] = s;
  }
#endif
  const v8f acc = a0 + a1;
  const float bv = bias ? bias[n0 + ln] : 0.f;
#pragma unroll
  for (int v = 0; v < 8; ++v)
    C[(size_t)(m0 + v + 8 * hh) * JJ + n0 + ln] = acc[v] + bv;
}

// ---------------------------------------------------------------------------
// Kernel 3: 2-layer LSTM, 64 sequential steps, B=4 padded to M=16.
// One workgroup (8 waves); recurrent matmuls via f16 WMMA, h-state f16 in LDS.
// ---------------------------------------------------------------------------
__global__ void __launch_bounds__(256) k_lstm(
    const float* __restrict__ G0x, const float* __restrict__ Whh0,
    const float* __restrict__ Wih1, const float* __restrict__ Whh1,
    const float* __restrict__ bih1, const float* __restrict__ bhh1,
    float* __restrict__ hdec_seq) {
  extern __shared__ char smem[];
  _Float16* hA0 = (_Float16*)smem;                 // [4][512] current h0 (f16)
  _Float16* hA1 = hA0 + BB * DU;                   // [4][512] current h1 (f16)
  float* c0    = (float*)(hA1 + BB * DU);          // [4][512]
  float* c1    = c0 + BB * DU;                     // [4][512]
  float* gates = c1 + BB * DU;                     // [4][2048]

  const int tid = threadIdx.x, lane = tid & 31, wave = tid >> 5;
  const int hh = lane >> 4, ln = lane & 15;
  const bool live_row = (ln < BB);                 // A rows >= 4 are zero pad

  for (int i = tid; i < BB * DU; i += 256) {
    hA0[i] = (_Float16)0.f; hA1[i] = (_Float16)0.f; c0[i] = 0.f; c1[i] = 0.f;
  }
  __syncthreads();

  for (int u = 0; u < UU; ++u) {
    // ---- layer 0: gates = h0 @ Whh0^T + G0x[:,u,:] ----
    for (int i = 0; i < 16; ++i) {
      const int n0 = (wave * 16 + i) * 16;
      const float* brow = Whh0 + (size_t)(n0 + ln) * DU;
      v8f a0 = {}, a1 = {};
#pragma unroll 4
      for (int ks = 0; ks < 16; ks += 2) {
        const int kb0 = ks * 32 + 8 * hh, kb1 = kb0 + 32;
        v16h x0 = {}, x1 = {};
        if (live_row) { x0 = frag_from_f16(hA0 + ln * DU + kb0);
                        x1 = frag_from_f16(hA0 + ln * DU + kb1); }
        v16h w0 = frag_from_f32(brow + kb0);
        v16h w1 = frag_from_f32(brow + kb1);
        a0 = __builtin_amdgcn_wmma_f32_16x16x32_f16(false, x0, false, w0, (short)0, a0, false, false);
        a1 = __builtin_amdgcn_wmma_f32_16x16x32_f16(false, x1, false, w1, (short)0, a1, false, false);
      }
      const v8f acc = a0 + a1;
      if (hh == 0) {                               // rows 0..3 live in lanes 0-15
#pragma unroll
        for (int v = 0; v < BB; ++v)
          gates[v * GG + n0 + ln] = acc[v] + G0x[(size_t)(v * UU + u) * GG + n0 + ln];
      }
    }
    __syncthreads();
    for (int idx = tid; idx < BB * DU; idx += 256) {   // cell 0 update
      const int b = idx >> 9, d = idx & 511;
      const float gi = gates[b * GG + d],        gf = gates[b * GG + 512 + d];
      const float gz = gates[b * GG + 1024 + d], go = gates[b * GG + 1536 + d];
      const float cc = sigf(gf) * c0[idx] + sigf(gi) * tanhf(gz);
      c0[idx] = cc;
      hA0[idx] = (_Float16)(sigf(go) * tanhf(cc));
    }
    __syncthreads();
    // ---- layer 1: gates = h0 @ Wih1^T + h1 @ Whh1^T + biases ----
    for (int i = 0; i < 16; ++i) {
      const int n0 = (wave * 16 + i) * 16;
      const float* bri = Wih1 + (size_t)(n0 + ln) * DU;
      const float* brh = Whh1 + (size_t)(n0 + ln) * DU;
      v8f a0 = {}, a1 = {};
#pragma unroll 4
      for (int ks = 0; ks < 16; ks += 2) {
        const int kb0 = ks * 32 + 8 * hh, kb1 = kb0 + 32;
        v16h x0 = {}, x1 = {};
        if (live_row) { x0 = frag_from_f16(hA0 + ln * DU + kb0);
                        x1 = frag_from_f16(hA0 + ln * DU + kb1); }
        v16h w0 = frag_from_f32(bri + kb0);
        v16h w1 = frag_from_f32(bri + kb1);
        a0 = __builtin_amdgcn_wmma_f32_16x16x32_f16(false, x0, false, w0, (short)0, a0, false, false);
        a1 = __builtin_amdgcn_wmma_f32_16x16x32_f16(false, x1, false, w1, (short)0, a1, false, false);
      }
#pragma unroll 4
      for (int ks = 0; ks < 16; ks += 2) {
        const int kb0 = ks * 32 + 8 * hh, kb1 = kb0 + 32;
        v16h x0 = {}, x1 = {};
        if (live_row) { x0 = frag_from_f16(hA1 + ln * DU + kb0);
                        x1 = frag_from_f16(hA1 + ln * DU + kb1); }
        v16h w0 = frag_from_f32(brh + kb0);
        v16h w1 = frag_from_f32(brh + kb1);
        a0 = __builtin_amdgcn_wmma_f32_16x16x32_f16(false, x0, false, w0, (short)0, a0, false, false);
        a1 = __builtin_amdgcn_wmma_f32_16x16x32_f16(false, x1, false, w1, (short)0, a1, false, false);
      }
      const v8f acc = a0 + a1;
      if (hh == 0) {
#pragma unroll
        for (int v = 0; v < BB; ++v)
          gates[v * GG + n0 + ln] = acc[v] + bih1[n0 + ln] + bhh1[n0 + ln];
      }
    }
    __syncthreads();
    for (int idx = tid; idx < BB * DU; idx += 256) {   // cell 1 update
      const int b = idx >> 9, d = idx & 511;
      const float gi = gates[b * GG + d],        gf = gates[b * GG + 512 + d];
      const float gz = gates[b * GG + 1024 + d], go = gates[b * GG + 1536 + d];
      const float cc = sigf(gf) * c1[idx] + sigf(gi) * tanhf(gz);
      c1[idx] = cc;
      const float h = sigf(go) * tanhf(cc);
      hA1[idx] = (_Float16)h;
      hdec_seq[(size_t)(b * UU + u) * DU + d] = h;
    }
    __syncthreads();
  }
}

// ---------------------------------------------------------------------------
// Kernel 4: fused joint: out[b,t,u,:] = tanh(henc[b,t,:]+hdecJ[b,u,:]) @ Wout^T + bout
// Per WG: fixed (b, 16-u tile, 32-t chunk, 128-n span). W_out fragments live in
// VGPRs for the whole kernel; A tile regenerated in LDS per t (tanh fused).
// hdec tile staged global->LDS via the Tensor Data Mover (TDM).
// ---------------------------------------------------------------------------
__global__ void __launch_bounds__(256) k_joint(
    const float* __restrict__ henc, const float* __restrict__ hdecJ,
    const float* __restrict__ Wout, const float* __restrict__ bout,
    float* __restrict__ out) {
  extern __shared__ char smem[];
  float* hdt = (float*)smem;                       // [16][512] f32 hdec tile
  _Float16* Atile = (_Float16*)(hdt + 16 * JJ);    // [16][512] f16 A tile

  const int tid = threadIdx.x, lane = tid & 31, wave = tid >> 5;
  const int hh = lane >> 4, ln = lane & 15;
  const int b  = blockIdx.y >> 2;
  const int u0 = (blockIdx.y & 3) * 16;
  const int n0 = blockIdx.x * 128 + wave * 16;
  const int t0 = blockIdx.z * 32;

  const float* src = hdecJ + (size_t)(b * UU + u0) * JJ;
#ifdef HAVE_TDM
  // TDM 2-D tile load: 512 x 16 fp32 tile, tensor stride 512 elems. D# per ISA 8.3/8.4.
  if (wave == 0) {
    const unsigned long long ga = (unsigned long long)(uintptr_t)src;
    const unsigned int ldsoff = (unsigned int)(uintptr_t)(void*)hdt;  // addr[31:0] = LDS offset
    v4u g0;
    g0[0] = 1u;                                    // count=1, user descriptor
    g0[1] = ldsoff;                                // lds_addr
    g0[2] = (unsigned int)ga;                      // global_addr[31:0]
    g0[3] = ((unsigned int)(ga >> 32) & 0x01FFFFFFu) | (2u << 30);  // addr[56:32] | type=2
    v8i g1;
    g1[0] = (int)(2u << 16);                       // workgroup_mask=0, data_size=4B
    g1[1] = (int)(512u << 16);                     // tensor_dim0 = 512 (low16 in [31:16])
    g1[2] = (int)(16u << 16);                      // tensor_dim0 hi=0 | tensor_dim1 = 16
    g1[3] = (int)(512u << 16);                     // tensor_dim1 hi=0 | tile_dim0 = 512
    g1[4] = 16;                                    // tile_dim1 = 16, tile_dim2 = 0
    g1[5] = 512;                                   // tensor_dim0_stride[31:0]
    g1[6] = 0;                                     // stride hi, tensor_dim1_stride
    g1[7] = 0;
    v4i g2 = {}, g3 = {};
#if __clang_major__ >= 23
    v8i g4 = {};
    __builtin_amdgcn_tensor_load_to_lds(g0, g1, g2, g3, g4, 0);
#else
    __builtin_amdgcn_tensor_load_to_lds(g0, g1, g2, g3, 0);
#endif
    __builtin_amdgcn_s_wait_tensorcnt(0);
  }
#else
  for (int i = tid; i < 16 * JJ; i += 256) hdt[i] = src[i];
#endif

  v16h Bf[16];                                     // W_out column block, in VGPRs
  {
    const float* brow = Wout + (size_t)(n0 + ln) * KK;
#pragma unroll
    for (int ks = 0; ks < 16; ++ks) Bf[ks] = frag_from_f32(brow + ks * 32 + 8 * hh);
  }
  const float bv = bout[n0 + ln];
  __syncthreads();

  for (int t = t0; t < t0 + 32; ++t) {
    const float* hrow = henc + (size_t)(b * TT + t) * JJ;
    for (int i = tid; i < 16 * JJ; i += 256) {     // build A = tanh(henc+hdec)
      const int k = i & 511;
      Atile[i] = (_Float16)tanhf(hrow[k] + hdt[i]);
    }
    if (t + 1 < TT) __builtin_prefetch(hrow + JJ, 0, 1);   // global_prefetch_b8
    __syncthreads();

    v8f a0 = {}, a1 = {};
#pragma unroll 8
    for (int ks = 0; ks < 16; ks += 2) {
      v16h x0 = frag_from_f16(Atile + ln * JJ + ks * 32 + 8 * hh);
      v16h x1 = frag_from_f16(Atile + ln * JJ + (ks + 1) * 32 + 8 * hh);
      a0 = __builtin_amdgcn_wmma_f32_16x16x32_f16(false, x0, false, Bf[ks], (short)0, a0, false, false);
      a1 = __builtin_amdgcn_wmma_f32_16x16x32_f16(false, x1, false, Bf[ks + 1], (short)0, a1, false, false);
    }
    const v8f acc = a0 + a1;
    float* orow = out + ((size_t)(b * TT + t) * UU + u0) * OD;
#pragma unroll
    for (int v = 0; v < 8; ++v)
      orow[(size_t)(v + 8 * hh) * OD + n0 + ln] = acc[v] + bv;
    __syncthreads();
  }
}

// ---------------------------------------------------------------------------
extern "C" void kernel_launch(void* const* d_in, const int* in_sizes, int n_in,
                              void* d_out, int out_size, void* d_ws, size_t ws_size,
                              hipStream_t stream) {
  (void)in_sizes; (void)n_in; (void)out_size; (void)ws_size;
  const float* hs_pad = (const float*)d_in[0];
  const int*   ys     = (const int*)d_in[2];
  const float* embed  = (const float*)d_in[3];
  const float* Wih0   = (const float*)d_in[4];
  const float* Whh0   = (const float*)d_in[5];
  const float* bih0   = (const float*)d_in[6];
  const float* bhh0   = (const float*)d_in[7];
  const float* Wih1   = (const float*)d_in[8];
  const float* Whh1   = (const float*)d_in[9];
  const float* bih1   = (const float*)d_in[10];
  const float* bhh1   = (const float*)d_in[11];
  const float* Wenc   = (const float*)d_in[12];
  const float* benc   = (const float*)d_in[13];
  const float* Wdec   = (const float*)d_in[14];
  const float* Wout   = (const float*)d_in[15];
  const float* bout   = (const float*)d_in[16];
  float* out = (float*)d_out;

  // workspace (~5 MB): G0x | hdec_seq | hdecJ | henc
  float* G0x      = (float*)d_ws;                       // 4*64*2048
  float* hdec_seq = G0x + (size_t)BB * UU * GG;         // 4*64*512
  float* hdecJ    = hdec_seq + (size_t)BB * UU * DU;    // 4*64*512
  float* henc     = hdecJ + (size_t)BB * UU * JJ;       // 4*256*512

  k_gates0<<<256, 256, 0, stream>>>(embed, ys, Wih0, bih0, bhh0, G0x);
  k_gemm_f32<<<256, 256, 0, stream>>>(hs_pad, Wenc, benc, henc, BB * TT);

  const size_t lstm_lds = 2 * (BB * DU) * 2    // hA0,hA1 f16
                        + 2 * (BB * DU) * 4    // c0,c1 f32
                        + (BB * GG) * 4;       // gates f32   = 56 KB
  k_lstm<<<1, 256, lstm_lds, stream>>>(G0x, Whh0, Wih1, Whh1, bih1, bhh1, hdec_seq);

  k_gemm_f32<<<64, 256, 0, stream>>>(hdec_seq, Wdec, nullptr, hdecJ, BB * UU);

  const size_t joint_lds = (size_t)16 * JJ * 4 + (size_t)16 * JJ * 2;  // 48 KB
  k_joint<<<dim3(8, 16, 8), 256, joint_lds, stream>>>(henc, hdecJ, Wout, bout, out);
}
